// MultiheadAttention_27693949125383
// MI455X (gfx1250) — matrix operand order repaired and checked by reading, not compiled, not run on hardware
//
#include <hip/hip_runtime.h>

// ---------------------------------------------------------------------------
// MultiheadAttention for MI455X (gfx1250), wave32 + v_wmma_f32_16x16x32_bf16.
//   [0] f32->bf16 weight convert
//   [1] x3 QKV projection GEMM: block stages its 16-row A tile in LDS
//       (f32->bf16 during copy), 8 waves cover all 512 N-cols, B fragments
//       double-buffered in registers (prefetch next K-step before WMMA).
//   [2] attention: K staged to LDS via global_load_async_to_lds_b128
//       (ASYNCcnt), V staged transposed; QK^T 1 WMMA/tile (K=dh=32),
//       +bias +mask[n%32], softmax, P via LDS, PV = 8x2 WMMAs.
//   [3] output projection GEMM (A bf16 from ws), f32 -> d_out.
// ---------------------------------------------------------------------------

#define C_EMB 512
#define HEADS 16
#define DH    32
#define LSEQ  256
#define NBATCH 128

#define USE_ASYNC_LDS 1

typedef __attribute__((ext_vector_type(16))) __bf16 v16bf;
typedef __attribute__((ext_vector_type(8)))  float  v8f;
typedef unsigned short u16;

__device__ __forceinline__ u16 f2bf(float f) {
  unsigned int u = __builtin_bit_cast(unsigned int, f);
  u += 0x7FFFu + ((u >> 16) & 1u);   // round-to-nearest-even
  return (u16)(u >> 16);
}

__device__ __forceinline__ unsigned lds_off(const void* p) {
  // generic pointer to LDS: low 32 bits are the LDS byte address
  return (unsigned)(uintptr_t)p;
}

union Frag16 { v16bf v; uint4 q[2]; u16 s[16]; };

// ---------------------------------------------------------------------------
// Kernel 0: convert in_proj_w (3*C*C) and out_proj_w (C*C) to bf16
// ---------------------------------------------------------------------------
__global__ void k_cvt_weights(const float* __restrict__ in_w,
                              const float* __restrict__ out_w,
                              u16* __restrict__ dst) {
  const int n_in4 = (3 * C_EMB * C_EMB) / 4;
  const int total4 = (4 * C_EMB * C_EMB) / 4;
  for (int i = blockIdx.x * blockDim.x + threadIdx.x; i < total4;
       i += gridDim.x * blockDim.x) {
    float4 f = (i < n_in4) ? ((const float4*)in_w)[i]
                           : ((const float4*)out_w)[i - n_in4];
    uint2 o;
    o.x = (unsigned)f2bf(f.x) | ((unsigned)f2bf(f.y) << 16);
    o.y = (unsigned)f2bf(f.z) | ((unsigned)f2bf(f.w) << 16);
    ((uint2*)dst)[i] = o;
  }
}

// ---------------------------------------------------------------------------
// GEMM: Y[32768 x 512] = A @ W^T (+bias).  Block = 256 thr (8 waves), each
// block owns 16 rows (contiguous in memory), staged once into LDS as bf16.
// Wave w covers N-cols [w*64, w*64+64) with 4 accumulators; B fragments are
// register double-buffered so loads for step k+1 issue before WMMAs of k.
// MODE 0: A is f32, output bf16 head-split (N,H,L,dh), (acc+bias)*scale.
// MODE 1: A is bf16, output f32 flat (M x 512), acc+bias.
// ---------------------------------------------------------------------------
template <int MODE>
__global__ __launch_bounds__(256) void k_gemm(
    const void* __restrict__ Aptr, const u16* __restrict__ Wbf,
    const float* __restrict__ bias, void* __restrict__ OutP, float scale) {
  __shared__ u16 Atile[16 * C_EMB];   // 16 KB
  const int m0 = blockIdx.x * 16;

  if constexpr (MODE == 0) {
    const float* src = (const float*)Aptr + (size_t)m0 * C_EMB;
    for (int i = threadIdx.x; i < (16 * C_EMB) / 4; i += 256) {
      float4 f = ((const float4*)src)[i];
      uint2 o;
      o.x = (unsigned)f2bf(f.x) | ((unsigned)f2bf(f.y) << 16);
      o.y = (unsigned)f2bf(f.z) | ((unsigned)f2bf(f.w) << 16);
      ((uint2*)Atile)[i] = o;
    }
  } else {
    const u16* src = (const u16*)Aptr + (size_t)m0 * C_EMB;
    for (int i = threadIdx.x; i < (16 * C_EMB) / 8; i += 256)
      ((uint4*)Atile)[i] = ((const uint4*)src)[i];
  }
  __syncthreads();

  const int wave = threadIdx.x >> 5;
  const int lane = threadIdx.x & 31;
  const int kg = lane >> 4, mr = lane & 15;
  const int n0 = wave * 64;

  v8f acc[4] = {};
  Frag16 bq[2][4];
  const u16* Wbase = Wbf + (size_t)mr * C_EMB + 16 * kg;

#pragma unroll
  for (int t = 0; t < 4; ++t) {
    const u16* w = Wbase + (size_t)(n0 + t * 16) * C_EMB;
    bq[0][t].q[0] = *(const uint4*)(w);
    bq[0][t].q[1] = *(const uint4*)(w + 8);
  }

#pragma unroll
  for (int kk = 0; kk < 16; ++kk) {
    const int cur = kk & 1, nxt = cur ^ 1;
    if (kk < 15) {                      // prefetch next K-step's B frags
      const int ks = (kk + 1) * 32;
#pragma unroll
      for (int t = 0; t < 4; ++t) {
        const u16* w = Wbase + (size_t)(n0 + t * 16) * C_EMB + ks;
        bq[nxt][t].q[0] = *(const uint4*)(w);
        bq[nxt][t].q[1] = *(const uint4*)(w + 8);
      }
    }
    Frag16 a;
    const u16* ar = Atile + mr * C_EMB + kk * 32;
    a.q[0] = *(const uint4*)(ar + 16 * kg);
    a.q[1] = *(const uint4*)(ar + 16 * kg + 8);
#pragma unroll
    for (int t = 0; t < 4; ++t)
      acc[t] = __builtin_amdgcn_wmma_f32_16x16x32_bf16(
          false, a.v, false, bq[cur][t].v, (short)0, acc[t], false, false);
  }

  const int nc = mr;
  if constexpr (MODE == 0) {
    u16* Obf = (u16*)OutP;
#pragma unroll
    for (int t = 0; t < 4; ++t) {
      const int n = n0 + t * 16 + nc;
      const int h = n >> 5, d = n & 31;
      const float bn = bias[n];
#pragma unroll
      for (int vv = 0; vv < 8; ++vv) {
        const int m = m0 + vv + 8 * kg;
        const int nb = m >> 8, l = m & 255;         // L = 256
        Obf[((((size_t)nb * HEADS + h) * LSEQ + l) * DH) + d] =
            f2bf((acc[t][vv] + bn) * scale);
      }
    }
  } else {
    float* Out = (float*)OutP;
#pragma unroll
    for (int t = 0; t < 4; ++t) {
      const int n = n0 + t * 16 + nc;
      const float bn = bias[n];
#pragma unroll
      for (int vv = 0; vv < 8; ++vv) {
        const int m = m0 + vv + 8 * kg;
        Out[(size_t)m * C_EMB + n] = acc[t][vv] + bn;
      }
    }
  }
}

// ---------------------------------------------------------------------------
// Kernel 2: attention core. Block = 128 thr (4 waves); wave owns 16 query rows.
// Grid: (L/64, H, N) = (4, 16, 128).  LDS: K row-major 16KB (async DMA),
// V transposed 16KB, P 32KB.
// ---------------------------------------------------------------------------
__global__ __launch_bounds__(128) void k_attn(
    const u16* __restrict__ Qbf, const u16* __restrict__ Kbf,
    const u16* __restrict__ Vbf, const float* __restrict__ bias,
    const float* __restrict__ mask, u16* __restrict__ Obf) {
  __shared__ u16 Klds[LSEQ * DH];          // K row-major (S x dh), 16 KB
  __shared__ u16 Vt[DH * LSEQ];            // V transposed (dh x S), 16 KB
  __shared__ u16 Plds[4 * 16 * LSEQ];      // per-wave P rows, 32 KB

  const int n = blockIdx.z, h = blockIdx.y;
  const int wave = threadIdx.x >> 5, lane = threadIdx.x & 31;
  const int kg = lane >> 4, ln = lane & 15;

  const size_t headOff = ((size_t)n * HEADS + h) * LSEQ * DH;
  const u16* Q = Qbf + headOff;
  const u16* K = Kbf + headOff;
  const u16* V = Vbf + headOff;

#if USE_ASYNC_LDS
  // Stage K (contiguous 16 KB) via async DMA to LDS, tracked by ASYNCcnt.
  {
    const unsigned kbase = lds_off(&Klds[0]);
    for (int i = threadIdx.x; i < (LSEQ * DH) / 8; i += blockDim.x) {
      const unsigned dst = kbase + i * 16;
      const u16* src = K + i * 8;
      asm volatile("global_load_async_to_lds_b128 %0, %1, off"
                   :: "v"(dst), "v"(src) : "memory");
    }
  }
#else
  for (int i = threadIdx.x; i < (LSEQ * DH) / 8; i += blockDim.x)
    ((uint4*)Klds)[i] = ((const uint4*)K)[i];
#endif

  // Stage V transposed into LDS (global reads coalesced).
  for (int idx = threadIdx.x; idx < LSEQ * DH; idx += blockDim.x) {
    const int s = idx >> 5, d = idx & 31;
    Vt[d * LSEQ + s] = V[idx];
  }
#if USE_ASYNC_LDS
  asm volatile("s_wait_asynccnt 0x0" ::: "memory");
#endif
  __syncthreads();

  const int m0 = blockIdx.x * 64 + wave * 16;

  // Q fragment (16x32, K = dh = 32): loaded once; Q pre-scaled by 1/sqrt(dh).
  Frag16 fq;
  {
    const u16* Qrow = Q + (size_t)(m0 + ln) * DH;
    fq.q[0] = *(const uint4*)(Qrow + 16 * kg);
    fq.q[1] = *(const uint4*)(Qrow + 16 * kg + 8);
  }

  // Scores: 16 tiles of 16x16, one WMMA each; then + bias + mask.
  const float* bptr = bias + (size_t)h * LSEQ * LSEQ;
  const float* mptr = mask + (size_t)(n & 31) * LSEQ * LSEQ;
  v8f sc[16];
#pragma unroll
  for (int st = 0; st < 16; ++st) {
    Frag16 fk;
    const u16* Krow = &Klds[(st * 16 + ln) * DH + 16 * kg];
    fk.q[0] = *(const uint4*)(Krow);
    fk.q[1] = *(const uint4*)(Krow + 8);
    v8f c = {};
    c = __builtin_amdgcn_wmma_f32_16x16x32_bf16(
        false, fq.v, false, fk.v, (short)0, c, false, false);
    const int s = st * 16 + ln;
#pragma unroll
    for (int vv = 0; vv < 8; ++vv) {
      const int m = m0 + vv + 8 * kg;
      c[vv] += bptr[(size_t)m * LSEQ + s] + mptr[(size_t)m * LSEQ + s];
    }
    sc[st] = c;
  }

  // Row softmax: row m = v + 8*kg lives across lanes 0..15 of each half.
  float rinv[8];
#pragma unroll
  for (int vv = 0; vv < 8; ++vv) {
    float mx = -1e30f;
#pragma unroll
    for (int st = 0; st < 16; ++st) mx = fmaxf(mx, sc[st][vv]);
#pragma unroll
    for (int off = 1; off < 16; off <<= 1) mx = fmaxf(mx, __shfl_xor(mx, off, 32));
    float sum = 0.f;
#pragma unroll
    for (int st = 0; st < 16; ++st) {
      const float e = __expf(sc[st][vv] - mx);
      sc[st][vv] = e;
      sum += e;
    }
#pragma unroll
    for (int off = 1; off < 16; off <<= 1) sum += __shfl_xor(sum, off, 32);
    rinv[vv] = 1.0f / sum;
  }

  // P (C-layout) -> LDS row-major bf16 so it can be re-read in A-layout.
  u16* Pw = &Plds[wave * 16 * LSEQ];
#pragma unroll
  for (int st = 0; st < 16; ++st)
#pragma unroll
    for (int vv = 0; vv < 8; ++vv)
      Pw[(vv + 8 * kg) * LSEQ + st * 16 + ln] = f2bf(sc[st][vv] * rinv[vv]);
  __syncthreads();

  // O = P (16x256) @ V (256x32): 8 K-chunks, 2 column tiles (dh = 32).
  v8f o0 = {}, o1 = {};
#pragma unroll
  for (int kk = 0; kk < 8; ++kk) {
    Frag16 fp;
    const u16* Pr = Pw + (size_t)ln * LSEQ + kk * 32;
    fp.q[0] = *(const uint4*)(Pr + 16 * kg);
    fp.q[1] = *(const uint4*)(Pr + 16 * kg + 8);
    Frag16 fv0, fv1;
    const u16* v0p = &Vt[(size_t)ln * LSEQ + kk * 32 + 16 * kg];
    const u16* v1p = &Vt[(size_t)(ln + 16) * LSEQ + kk * 32 + 16 * kg];
    fv0.q[0] = *(const uint4*)(v0p);
    fv0.q[1] = *(const uint4*)(v0p + 8);
    fv1.q[0] = *(const uint4*)(v1p);
    fv1.q[1] = *(const uint4*)(v1p + 8);
    o0 = __builtin_amdgcn_wmma_f32_16x16x32_bf16(
        false, fp.v, false, fv0.v, (short)0, o0, false, false);
    o1 = __builtin_amdgcn_wmma_f32_16x16x32_bf16(
        false, fp.v, false, fv1.v, (short)0, o1, false, false);
  }

  // Write head-merged bf16 output: (N, L, C) columns h*32 + d.
  const size_t obase = ((size_t)n * LSEQ) * C_EMB + (size_t)h * DH;
#pragma unroll
  for (int vv = 0; vv < 8; ++vv) {
    const int m = m0 + vv + 8 * kg;
    u16* orow = Obf + obase + (size_t)m * C_EMB;
    orow[ln] = f2bf(o0[vv]);
    orow[ln + 16] = f2bf(o1[vv]);
  }
}

// ---------------------------------------------------------------------------
// Launch
// ---------------------------------------------------------------------------
extern "C" void kernel_launch(void* const* d_in, const int* in_sizes, int n_in,
                              void* d_out, int out_size, void* d_ws,
                              size_t ws_size, hipStream_t stream) {
  const float* q     = (const float*)d_in[0];
  const float* k     = (const float*)d_in[1];
  const float* v     = (const float*)d_in[2];
  const float* b     = (const float*)d_in[3];
  const float* mask  = (const float*)d_in[4];
  const float* in_w  = (const float*)d_in[5];
  const float* in_b  = (const float*)d_in[6];
  const float* out_w = (const float*)d_in[7];
  const float* out_b = (const float*)d_in[8];
  float* out = (float*)d_out;

  const size_t CC  = (size_t)C_EMB * C_EMB;           // 262144
  const size_t NLC = (size_t)NBATCH * LSEQ * C_EMB;   // 16777216

  u16* ws = (u16*)d_ws;
  u16* w_bf    = ws;                 // [0,3CC): wq|wk|wv, [3CC,4CC): wout
  u16* wout_bf = ws + 3 * CC;
  u16* q_bf    = ws + 4 * CC;
  u16* k_bf    = q_bf + NLC;
  u16* v_bf    = k_bf + NLC;
  u16* a_bf    = v_bf + NLC;

  // [0] weight conversion
  k_cvt_weights<<<1024, 256, 0, stream>>>(in_w, out_w, w_bf);

  // [1] QKV projections (1/sqrt(32) folded into Q)
  const float qscale = 0.17677669529663688f;
  k_gemm<0><<<2048, 256, 0, stream>>>(q, w_bf,          in_b,             q_bf, qscale);
  k_gemm<0><<<2048, 256, 0, stream>>>(k, w_bf + CC,     in_b + C_EMB,     k_bf, 1.0f);
  k_gemm<0><<<2048, 256, 0, stream>>>(v, w_bf + 2 * CC, in_b + 2 * C_EMB, v_bf, 1.0f);

  // [2] attention
  dim3 gattn(LSEQ / 64, HEADS, NBATCH);
  k_attn<<<gattn, 128, 0, stream>>>(q_bf, k_bf, v_bf, b, mask, a_bf);

  // [3] output projection
  k_gemm<1><<<2048, 256, 0, stream>>>(a_bf, wout_bf, out_b, out, 1.0f);
}
